// SpikeMLP_86397562126872
// MI455X (gfx1250) — compile-verified, matrix operand order checked by reading, not run on hardware
//
#include <hip/hip_runtime.h>
#include <hip/hip_bf16.h>
#include <stdint.h>

// Problem sizes (fixed by the reference)
#define BB 16
#define TT 4
#define SS 256
#define DD 512
#define HH 2048
#define RS (BB*SS)            // 4096 (b,s) pairs
#define MROWS (BB*TT*SS)      // 16384 GEMM rows
#define KC 128                // K-chunk staged into LDS in GEMM2

typedef __attribute__((ext_vector_type(16))) __bf16 v16bf;
typedef __attribute__((ext_vector_type(8)))  __bf16 bf16x8;
typedef __attribute__((ext_vector_type(4)))  __bf16 bf16x4;
typedef __attribute__((ext_vector_type(8)))  float  v8f;
typedef __attribute__((ext_vector_type(4)))  float  f32x4;
typedef __attribute__((ext_vector_type(4)))  unsigned u32x4;
typedef __attribute__((ext_vector_type(8)))  int     i32x8;
typedef __attribute__((ext_vector_type(4)))  int     i32x4;

// ---------------------------------------------------------------- fp32 -> bf16
__global__ __launch_bounds__(256) void k_cvt_bf16(const float* __restrict__ src,
                                                  __hip_bfloat16* __restrict__ dst,
                                                  int n)
{
  int i = (blockIdx.x * 256 + threadIdx.x) * 4;
  if (i < n) {
    f32x4 f = *(const f32x4*)(src + i);
    bf16x4 o;
    o[0] = (__bf16)f[0]; o[1] = (__bf16)f[1];
    o[2] = (__bf16)f[2]; o[3] = (__bf16)f[3];
    *(bf16x4*)(dst + i) = o;
  }
}

// ---------------------------------------------------------------- helpers
__global__ void k_zero(unsigned* c) { *c = 0u; }

__global__ void k_rate(const unsigned* __restrict__ c, float* __restrict__ rate)
{
  // rate = spikes.mean() over 16*4*256*2048 = 33554432 elements
  *rate = (float)(*c) * (1.0f / 33554432.0f);
}

// Build an A/B fragment (ISA 16-bit layout: lane holds K {kb..kb+7, kb+16..kb+23})
__device__ __forceinline__ v16bf make_frag(const __bf16* p)
{
  bf16x8 lo = *(const bf16x8*)p;
  bf16x8 hi = *(const bf16x8*)(p + 16);
  v16bf f;
#pragma unroll
  for (int j = 0; j < 8; ++j) { f[j] = lo[j]; f[j + 8] = hi[j]; }
  return f;
}

// ---------------------------------------------------------------- GEMM1 + LIF fused
// One wave: 16 (b,s)-rows x 32 hidden cols, all 4 time steps (8 f32 accumulators).
// A fragment is reused across 2 hidden tiles, B fragments across 4 time steps (8 WMMAs / 12 loads).
// Grid = (RS/16, HH/256), block = 8 waves.
__global__ __launch_bounds__(256) void k_fc1_lif(const __hip_bfloat16* __restrict__ xb_,
                                                 const __hip_bfloat16* __restrict__ w1b_,
                                                 __hip_bfloat16* __restrict__ spk,
                                                 unsigned* __restrict__ cnt)
{
  const __bf16* xb  = (const __bf16*)xb_;
  const __bf16* w1b = (const __bf16*)w1b_;

  const int lane = threadIdx.x & 31;
  const int wave = threadIdx.x >> 5;
  const int h0   = (blockIdx.y * 8 + wave) * 32;   // hidden tile base (2x16)
  const int rs0  = blockIdx.x * 16;                // (b,s) tile base; never crosses b
  const int b    = rs0 / SS;
  const int s0   = rs0 - b * SS;

  const int mrow = lane & 15;   // A row / B column within a 16-tile
  const int sel  = lane >> 4;   // K-half select (ISA 16-bit A/B layout)

  const __bf16* aB[TT];
#pragma unroll
  for (int t = 0; t < TT; ++t)
    aB[t] = xb + (size_t)((b * TT + t) * SS + s0 + mrow) * DD + sel * 8;
  const __bf16* bB0 = w1b + (size_t)(h0 + mrow) * DD + sel * 8;
  const __bf16* bB1 = bB0 + (size_t)16 * DD;

  v8f acc[TT][2] = {};

  for (int k = 0; k < DD; k += 32) {
    const v16bf bf0 = make_frag(bB0 + k);
    const v16bf bf1 = make_frag(bB1 + k);
#pragma unroll
    for (int t = 0; t < TT; ++t) {
      const v16bf af = make_frag(aB[t] + k);
      acc[t][0] = __builtin_amdgcn_wmma_f32_16x16x32_bf16(
          false, af, false, bf0, (short)0, acc[t][0], false, false);
      acc[t][1] = __builtin_amdgcn_wmma_f32_16x16x32_bf16(
          false, af, false, bf1, (short)0, acc[t][1], false, false);
    }
  }

  // ---- LIF recurrence in registers: v = beta*v + h_t; spike; hard reset ----
  const float beta = 1.0f - 1.0f / 20.0f;   // TAU_MEM = 20
  const int cm = sel * 8;                   // C layout: M = i + 8*(lane>=16)
  const int cn = lane & 15;                 // C layout: N = lane & 15
  unsigned my = 0;
#pragma unroll
  for (int nt = 0; nt < 2; ++nt) {
#pragma unroll
    for (int i = 0; i < 8; ++i) {
      float v = 0.0f;
#pragma unroll
      for (int t = 0; t < TT; ++t) {
        v = beta * v + acc[t][nt][i];
        const bool fired = v > 1.0f;        // VTH = 1.0
        my += fired ? 1u : 0u;
        const size_t r = (size_t)((b * TT + t) * SS + s0 + cm + i);
        spk[r * HH + h0 + nt * 16 + cn] = __float2bfloat16(fired ? 1.0f : 0.0f);
        v = fired ? 0.0f : v;               // hard reset
      }
    }
  }

  // spike-count reduction for the rate scalar
  __shared__ unsigned bc;
  if (threadIdx.x == 0) bc = 0u;
  __syncthreads();
  atomicAdd(&bc, my);
  __syncthreads();
  if (threadIdx.x == 0) atomicAdd(cnt, bc);
}

// ---------------------------------------------------------------- TDM tile load
// Stage a 16 x KC bf16 tile of W2 (row stride HH elements) into LDS via the
// Tensor Data Mover (CDNA5 ISA descriptor layout).
#if __has_builtin(__builtin_amdgcn_tensor_load_to_lds)
__device__ __forceinline__ void tdm_load_w2_tile(const __hip_bfloat16* gsrc, void* ldst)
{
  const uint64_t ga = (uint64_t)(uintptr_t)gsrc;
  const unsigned lo = (unsigned)(uintptr_t)ldst;    // flat LDS addr: low 32 = LDS offset
  u32x4 g0;
  g0[0] = 1u;                                       // count=1 (valid), user mode
  g0[1] = lo;                                       // lds_addr (bytes)
  g0[2] = (unsigned)ga;                             // global_addr[31:0]
  g0[3] = (unsigned)((ga >> 32) & 0x01FFFFFFu) | 0x80000000u; // addr[56:32] | type=2
  const uint64_t q0 = (1ull << 16)                           // data_size = 2 bytes
                    | ((uint64_t)(HH & 0xFFFF) << 48);       // tensor_dim0 lo16 = 2048
  const uint64_t q1 = (uint64_t)(HH >> 16)                   // tensor_dim0 hi16
                    | ((uint64_t)DD << 16)                   // tensor_dim1 = 512 rows
                    | ((uint64_t)KC << 48);                  // tile_dim0 = 128 elems
  const uint64_t q2 = 16ull                                  // tile_dim1 = 16 rows
                    | ((uint64_t)(unsigned)HH << 32);        // tensor_dim0_stride lo32
  const uint64_t q3 = 0ull;
  i32x8 g1;
  g1[0] = (int)(unsigned)q0; g1[1] = (int)(q0 >> 32);
  g1[2] = (int)(unsigned)q1; g1[3] = (int)(q1 >> 32);
  g1[4] = (int)(unsigned)q2; g1[5] = (int)(q2 >> 32);
  g1[6] = (int)(unsigned)q3; g1[7] = (int)(q3 >> 32);
  i32x4 z4 = {};
#if defined(__clang_major__) && (__clang_major__ >= 23)
  i32x8 z8 = {};
  __builtin_amdgcn_tensor_load_to_lds(g0, g1, z4, z4, z8, 0);
#else
  __builtin_amdgcn_tensor_load_to_lds(g0, g1, z4, z4, 0);
#endif
}
#endif

// ---------------------------------------------------------------- GEMM2 (spikes x W2^T)
// Block = 8 waves, each 32 rows x 16 cols (B fragment reused by 2 WMMAs).
// W2 tiles double-buffered in LDS via TDM: issue next, s_wait_tensorcnt(1).
// Grid = (MROWS/256, DD/16).
__global__ __launch_bounds__(256) void k_fc2(const __hip_bfloat16* __restrict__ spk_,
                                             const __hip_bfloat16* __restrict__ w2b_,
                                             float* __restrict__ out)
{
  __shared__ __hip_bfloat16 ldsB[2][16 * KC];   // 2 x 4 KB
  const __bf16* spk = (const __bf16*)spk_;

  const int lane = threadIdx.x & 31;
  const int wave = threadIdx.x >> 5;
  const int n0   = blockIdx.y * 16;                  // output-column tile
  const int r0   = blockIdx.x * 256 + wave * 32;     // 2 row tiles per wave
  const int mrow = lane & 15;
  const int sel  = lane >> 4;

  const __bf16* aB0 = spk + (size_t)(r0 + mrow) * HH + sel * 8;
  const __bf16* aB1 = aB0 + (size_t)16 * HH;
  const __hip_bfloat16* w2row = w2b_ + (size_t)n0 * HH;

  v8f acc0 = {}, acc1 = {};

#if __has_builtin(__builtin_amdgcn_tensor_load_to_lds)
  if (threadIdx.x == 0) tdm_load_w2_tile(w2row, &ldsB[0][0]);   // prime buffer 0
#endif

  for (int kc = 0; kc < HH; kc += KC) {
    const int buf = (kc / KC) & 1;
#if __has_builtin(__builtin_amdgcn_tensor_load_to_lds)
    if (threadIdx.x == 0) {
      if (kc + KC < HH) {
        tdm_load_w2_tile(w2row + kc + KC, &ldsB[buf ^ 1][0]);   // overlap next DMA
        __builtin_amdgcn_s_wait_tensorcnt(1);                   // current tile done
      } else {
        __builtin_amdgcn_s_wait_tensorcnt(0);
      }
    }
#else
    { // cooperative fallback: 256 threads x 16B = 4KB tile
      const int idx = threadIdx.x * 8;
      const int rr = idx / KC, cc = idx - rr * KC;
      *(bf16x8*)((__bf16*)&ldsB[buf][0] + idx) =
          *(const bf16x8*)((const __bf16*)w2row + (size_t)rr * HH + kc + cc);
    }
#endif
    if (kc + KC < HH) __builtin_prefetch(aB0 + kc + KC, 0, 1);  // global_prefetch_b8
    __syncthreads();

    const __bf16* lbase = (const __bf16*)&ldsB[buf][0] + mrow * KC + sel * 8;
#pragma unroll
    for (int kk = 0; kk < KC; kk += 32) {
      const v16bf bf  = make_frag(lbase + kk);      // ds_load_b128 x2
      const v16bf af0 = make_frag(aB0 + kc + kk);
      const v16bf af1 = make_frag(aB1 + kc + kk);
      acc0 = __builtin_amdgcn_wmma_f32_16x16x32_bf16(
          false, af0, false, bf, (short)0, acc0, false, false);
      acc1 = __builtin_amdgcn_wmma_f32_16x16x32_bf16(
          false, af1, false, bf, (short)0, acc1, false, false);
    }
    __syncthreads();
  }

  const int cm = sel * 8;
  const int cn = lane & 15;
#pragma unroll
  for (int i = 0; i < 8; ++i) {
    out[(size_t)(r0 + cm + i) * DD + n0 + cn]      = acc0[i];
    out[(size_t)(r0 + 16 + cm + i) * DD + n0 + cn] = acc1[i];
  }
}

// ---------------------------------------------------------------- launch
extern "C" void kernel_launch(void* const* d_in, const int* in_sizes, int n_in,
                              void* d_out, int out_size, void* d_ws, size_t ws_size,
                              hipStream_t stream)
{
  (void)in_sizes; (void)n_in; (void)out_size; (void)ws_size;
  const float* x  = (const float*)d_in[0];   // (16,4,256,512)
  const float* w1 = (const float*)d_in[1];   // (2048,512)
  const float* w2 = (const float*)d_in[2];   // (512,2048)
  float* out = (float*)d_out;                // (16,4,256,512) + [rate]

  char* ws = (char*)d_ws;
  unsigned*       cnt = (unsigned*)ws;                         // spike counter
  __hip_bfloat16* w1b = (__hip_bfloat16*)(ws + 256);           // 2 MB
  __hip_bfloat16* w2b = w1b + (size_t)HH * DD;                 // 2 MB
  __hip_bfloat16* xb  = w2b + (size_t)DD * HH;                 // 16 MB
  __hip_bfloat16* spk = xb  + (size_t)MROWS * DD;              // 64 MB

  k_cvt_bf16<<<dim3((MROWS * DD / 4 + 255) / 256), dim3(256), 0, stream>>>(x,  xb,  MROWS * DD);
  k_cvt_bf16<<<dim3((HH * DD / 4 + 255) / 256),    dim3(256), 0, stream>>>(w1, w1b, HH * DD);
  k_cvt_bf16<<<dim3((DD * HH / 4 + 255) / 256),    dim3(256), 0, stream>>>(w2, w2b, DD * HH);
  k_zero<<<dim3(1), dim3(1), 0, stream>>>(cnt);
  k_fc1_lif<<<dim3(RS / 16, HH / 256), dim3(256), 0, stream>>>(xb, w1b, spk, cnt);
  k_fc2<<<dim3(MROWS / 256, DD / 16), dim3(256), 0, stream>>>(spk, w2b, out);
  k_rate<<<dim3(1), dim3(1), 0, stream>>>(cnt, out + (size_t)MROWS * DD);
}
